// MultiHeadedAttention_75076028334674
// MI455X (gfx1250) — matrix operand order repaired and checked by reading, not compiled, not run on hardware
//
#include <hip/hip_runtime.h>
#include <hip/hip_bf16.h>
#include <math.h>
#include <stdint.h>

// ---------------------------------------------------------------------------
// MHA for MI455X (gfx1250): f16 WMMA (16x16x32, fp32 accum) for all GEMMs.
// Projection GEMMs: 8-wave workgroups, 128x64 tiles, B staged in LDS via
// async global->LDS copies (ASYNCcnt), double-buffered.
// Attention: flash-style online softmax, one wave per (b,h,16-query tile).
// Fragment layouts per CDNA5 ISA 7.12.2 (wave32).
// ---------------------------------------------------------------------------

#define BATCH 16
#define SEQ   512
#define DMODEL 1024
#define NHEAD 16
#define HDIM  64

typedef _Float16 v16h __attribute__((ext_vector_type(16)));
typedef _Float16 v8h  __attribute__((ext_vector_type(8)));
typedef float    v8f  __attribute__((ext_vector_type(8)));

// Build a 16-half A/B fragment from two 16-byte contiguous loads.
__device__ __forceinline__ v16h load_frag16(const _Float16* p0, const _Float16* p1) {
    v8h a = *(const v8h*)p0;
    v8h b = *(const v8h*)p1;
    v16h r;
#pragma unroll
    for (int i = 0; i < 8; ++i) { r[i] = a[i]; r[i + 8] = b[i]; }
    return r;
}

// Async global -> LDS copy, 16 bytes per lane, tracked on ASYNCcnt.
// vdst VGPR holds the LDS byte address (generic shared ptr truncated to 32b),
// vaddr pair holds the 64-bit global address (GV mode).
__device__ __forceinline__ void async_ld_b128(uint32_t lds_addr, const _Float16* gptr) {
    asm volatile("global_load_async_to_lds_b128 %0, %1, off"
                 :: "v"(lds_addr), "v"((uint64_t)(uintptr_t)gptr)
                 : "memory");
}
__device__ __forceinline__ void wait_async0() {
    asm volatile("s_wait_asynccnt 0x0" ::: "memory");
}

// Reductions across the 16 lanes of a half-wave group (rows of a C tile).
__device__ __forceinline__ float redmax16(float v) {
    v = fmaxf(v, __shfl_xor(v, 1, 32));
    v = fmaxf(v, __shfl_xor(v, 2, 32));
    v = fmaxf(v, __shfl_xor(v, 4, 32));
    v = fmaxf(v, __shfl_xor(v, 8, 32));
    return v;
}
__device__ __forceinline__ float redsum16(float v) {
    v += __shfl_xor(v, 1, 32);
    v += __shfl_xor(v, 2, 32);
    v += __shfl_xor(v, 4, 32);
    v += __shfl_xor(v, 8, 32);
    return v;
}

// ---------------- conversion kernels ----------------

__global__ void mha_cvt_x(const float* __restrict__ in, _Float16* __restrict__ out, int n) {
    int i = blockIdx.x * 256 + threadIdx.x;
    if (i < n) out[i] = (_Float16)in[i];
}

// W is [in, out] row-major fp32; produce WT [out, in] row-major f16.
__global__ void mha_cvt_wT(const float* __restrict__ Wq, const float* __restrict__ Wk,
                           const float* __restrict__ Wv, const float* __restrict__ Wo,
                           _Float16* __restrict__ WqT, _Float16* __restrict__ WkT,
                           _Float16* __restrict__ WvT, _Float16* __restrict__ WoT) {
    int w = blockIdx.y;
    const float* W  = (w == 0) ? Wq : (w == 1) ? Wk : (w == 2) ? Wv : Wo;
    _Float16*    WT = (w == 0) ? WqT : (w == 1) ? WkT : (w == 2) ? WvT : WoT;
    int idx = blockIdx.x * 256 + threadIdx.x;   // 0 .. 1048575, coalesced read
    int o = idx & (DMODEL - 1);
    int i = idx >> 10;
    WT[(size_t)o * DMODEL + i] = (_Float16)W[idx];
}

// ---------------- staged GEMM (128M x 64N per 8-wave workgroup) -------------
// MODE 0: QKV projection (blockIdx.z selects Q/K/V; f16 outputs, V transposed)
// MODE 1: output projection (fp32 output + bias)
template <int MODE>
__global__ void __launch_bounds__(256)
mha_gemm(const _Float16* __restrict__ Ah,
         const _Float16* __restrict__ WT0, const _Float16* __restrict__ WT1,
         const _Float16* __restrict__ WT2,
         const float* __restrict__ bias0, const float* __restrict__ bias1,
         const float* __restrict__ bias2,
         _Float16* __restrict__ Qh, _Float16* __restrict__ Kh,
         _Float16* __restrict__ VTh, float* __restrict__ Of) {
    const int which = (MODE == 0) ? blockIdx.z : 0;
    const _Float16* WT   = (MODE == 1) ? WT0
                         : (which == 0) ? WT0 : (which == 1) ? WT1 : WT2;
    const float*    bias = (MODE == 1) ? bias0
                         : (which == 0) ? bias0 : (which == 1) ? bias1 : bias2;

    __shared__ __align__(16) _Float16 Bbuf[2][64 * 32];   // 2 x 4KB double buffer

    const int t = threadIdx.x;
    const int w = t >> 5, lane = t & 31;
    const int g = lane >> 4, m = lane & 15;
    const int mBase = blockIdx.x * 128 + w * 16;
    const int nBase = blockIdx.y * 64;

    // Staging map: thread t copies 8 halves of BT row (nBase + t/4).
    const int sn = t >> 2;           // 0..63 local N
    const int sk = (t & 3) * 8;      // 0,8,16,24 within 32-wide K chunk
    const _Float16* Bsrc = WT + (size_t)(nBase + sn) * DMODEL + sk;

    async_ld_b128((uint32_t)(uintptr_t)&Bbuf[0][sn * 32 + sk], Bsrc);

    v8f acc[4] = {};
    const _Float16* Arow = Ah + (size_t)(mBase + m) * DMODEL;

    for (int c = 0; c < DMODEL / 32; ++c) {
        const int p = c & 1;
        wait_async0();            // this wave's copy done
        __syncthreads();          // everyone's copy done -> buf[p] valid
        if (c + 1 < DMODEL / 32)  // prefetch next chunk into the other buffer
            async_ld_b128((uint32_t)(uintptr_t)&Bbuf[p ^ 1][sn * 32 + sk],
                          Bsrc + (size_t)(c + 1) * 32);

        const int kk = c * 32;
        v16h a = load_frag16(Arow + kk + 8 * g, Arow + kk + 16 + 8 * g);
#pragma unroll
        for (int t4 = 0; t4 < 4; ++t4) {
            const _Float16* Bl = &Bbuf[p][(t4 * 16 + m) * 32 + 8 * g];
            v16h b = load_frag16(Bl, Bl + 16);
            acc[t4] = __builtin_amdgcn_wmma_f32_16x16x32_f16(
                false, a, false, b, (short)0, acc[t4], false, false);
        }
        __syncthreads();          // reads of buf[p] done before it is refilled
    }

#pragma unroll
    for (int t4 = 0; t4 < 4; ++t4) {
        const int col = nBase + t4 * 16 + m;       // lanes carry columns
        const float bv_ = bias[col];
        const int h = col >> 6, d = col & 63;
#pragma unroll
        for (int r = 0; r < 8; ++r) {
            const int row = mBase + r + 8 * g;
            const float v = acc[t4][r] + bv_;
            if (MODE == 1) {
                Of[(size_t)row * DMODEL + col] = v;
            } else {
                const int b = row >> 9, s = row & (SEQ - 1);
                if (which == 0)
                    Qh[(((size_t)(b * NHEAD + h) * SEQ + s) * HDIM) + d] = (_Float16)v;
                else if (which == 1)
                    Kh[(((size_t)(b * NHEAD + h) * SEQ + s) * HDIM) + d] = (_Float16)v;
                else
                    VTh[(((size_t)(b * NHEAD + h) * HDIM + d) * SEQ) + s] = (_Float16)v;
            }
        }
    }
}

// ---------------- RoPE (interleaved-pair) on Q and K ----------------
__global__ void mha_rope(_Float16* __restrict__ Qh, _Float16* __restrict__ Kh) {
    _Float16* P = blockIdx.y ? Kh : Qh;
    size_t idx = (size_t)blockIdx.x * 256 + threadIdx.x;  // B*H*S*32 pairs
    int i = (int)(idx & 31);                               // pair index in head
    size_t rest = idx >> 5;                                // (b*H + h)*S + s
    int s = (int)(rest & (SEQ - 1));
    size_t off = rest * HDIM + 2 * i;
    float x1 = (float)P[off], x2 = (float)P[off + 1];
    // freq = s * base^(-2i/64);  ln(10000) = 9.210340371976184
    float f = (float)s * expf(-((float)(2 * i) / 64.0f) * 9.210340371976184f);
    float sn, cs;
    sincosf(f, &sn, &cs);
    P[off]     = (_Float16)(x1 * cs - x2 * sn);
    P[off + 1] = (_Float16)(x1 * sn + x2 * cs);
}

// ---------------- flash attention: one wave per (b,h, 16-query tile) --------
__global__ void __launch_bounds__(32)
mha_attn(const _Float16* __restrict__ Qh, const _Float16* __restrict__ Kh,
         const _Float16* __restrict__ VTh, _Float16* __restrict__ ctx) {
    __shared__ __align__(16) _Float16 Pt[16 * 32];   // C->A relayout bounce

    const int qt = blockIdx.x;            // 0..31 query tile
    const int bh = blockIdx.y;            // 0..255 = b*NHEAD + h
    const int lane = threadIdx.x;
    const int g = lane >> 4, m = lane & 15;
    const int q0 = qt * 16;

    const _Float16* Qb = Qh  + (size_t)bh * SEQ * HDIM;
    const _Float16* Kb = Kh  + (size_t)bh * SEQ * HDIM;
    const _Float16* Vb = VTh + (size_t)bh * HDIM * SEQ;

    // Resident Q fragments covering HD=64 (two 16x32 A tiles).
    const _Float16* Qrow = Qb + (size_t)(q0 + m) * HDIM;
    const v16h qa0 = load_frag16(Qrow + 8 * g,      Qrow + 16 + 8 * g);
    const v16h qa1 = load_frag16(Qrow + 32 + 8 * g, Qrow + 48 + 8 * g);

    v8f o[4] = {};
    float Mr[8], Lr[8];
#pragma unroll
    for (int r = 0; r < 8; ++r) { Mr[r] = -__builtin_inff(); Lr[r] = 0.0f; }

    const int nkb = qt / 2 + 1;           // causal: only key blocks <= diag
    for (int kb = 0; kb < nkb; ++kb) {
        const int n0 = kb * 32;

        // ---- scores: two 16x16 tiles, K-dim = HD = 64 (2 WMMAs each) ----
        v8f s0 = {}, s1 = {};
        {
            const _Float16* Kr = Kb + (size_t)(n0 + m) * HDIM;
            v16h b0 = load_frag16(Kr + 8 * g,      Kr + 16 + 8 * g);
            v16h b1 = load_frag16(Kr + 32 + 8 * g, Kr + 48 + 8 * g);
            s0 = __builtin_amdgcn_wmma_f32_16x16x32_f16(false, qa0, false, b0, (short)0, s0, false, false);
            s0 = __builtin_amdgcn_wmma_f32_16x16x32_f16(false, qa1, false, b1, (short)0, s0, false, false);
        }
        {
            const _Float16* Kr = Kb + (size_t)(n0 + 16 + m) * HDIM;
            v16h b0 = load_frag16(Kr + 8 * g,      Kr + 16 + 8 * g);
            v16h b1 = load_frag16(Kr + 32 + 8 * g, Kr + 48 + 8 * g);
            s1 = __builtin_amdgcn_wmma_f32_16x16x32_f16(false, qa0, false, b0, (short)0, s1, false, false);
            s1 = __builtin_amdgcn_wmma_f32_16x16x32_f16(false, qa1, false, b1, (short)0, s1, false, false);
        }

        // ---- online softmax over the 32-key block ----
#pragma unroll
        for (int r = 0; r < 8; ++r) {
            const int qrow = q0 + r + 8 * g;
            float a = s0[r] * 0.125f;                 // 1/sqrt(64)
            float bb = s1[r] * 0.125f;
            if (n0 + m > qrow)      a  = -__builtin_inff();
            if (n0 + 16 + m > qrow) bb = -__builtin_inff();
            float mx = redmax16(fmaxf(a, bb));
            float Mn = fmaxf(Mr[r], mx);
            float alpha = __expf(Mr[r] - Mn);
            float pa = __expf(a - Mn);
            float pb = __expf(bb - Mn);
            float sum = redsum16(pa + pb);
            Lr[r] = Lr[r] * alpha + sum;
            Mr[r] = Mn;
            o[0][r] *= alpha; o[1][r] *= alpha; o[2][r] *= alpha; o[3][r] *= alpha;
            Pt[(r + 8 * g) * 32 + m]      = (_Float16)pa;
            Pt[(r + 8 * g) * 32 + 16 + m] = (_Float16)pb;
        }
        __syncthreads();

        // ---- O += P(16x32) @ V(32x64): 4 WMMAs, B from transposed V ----
        const _Float16* Pr = &Pt[m * 32];
        v16h pfrag = load_frag16(Pr + 8 * g, Pr + 16 + 8 * g);
#pragma unroll
        for (int t = 0; t < 4; ++t) {
            const _Float16* Vr = Vb + (size_t)(t * 16 + m) * SEQ + n0 + 8 * g;
            v16h bf = load_frag16(Vr, Vr + 16);
            o[t] = __builtin_amdgcn_wmma_f32_16x16x32_f16(
                false, pfrag, false, bf, (short)0, o[t], false, false);
        }
        __syncthreads();
    }

    // ---- normalize and store context as [B, S, H*HD] f16 ----
    const int b = bh >> 4, h = bh & 15;
#pragma unroll
    for (int t = 0; t < 4; ++t) {
#pragma unroll
        for (int r = 0; r < 8; ++r) {
            const int s = q0 + r + 8 * g;
            const float v = o[t][r] / Lr[r];
            ctx[((size_t)(b * SEQ + s) * NHEAD + h) * HDIM + t * 16 + m] = (_Float16)v;
        }
    }
}

// ---------------------------------------------------------------------------

extern "C" void kernel_launch(void* const* d_in, const int* in_sizes, int n_in,
                              void* d_out, int out_size, void* d_ws, size_t ws_size,
                              hipStream_t stream) {
    const float* x  = (const float*)d_in[0];
    const float* Wq = (const float*)d_in[1];
    const float* bq = (const float*)d_in[2];
    const float* Wk = (const float*)d_in[3];
    const float* bk = (const float*)d_in[4];
    const float* Wv = (const float*)d_in[5];
    const float* bv = (const float*)d_in[6];
    const float* Wo = (const float*)d_in[7];
    const float* bo = (const float*)d_in[8];
    float* out = (float*)d_out;

    // Workspace carve-up (all 256B aligned): ~93 MB total.
    char* ws = (char*)d_ws;
    const size_t SZ_X = (size_t)BATCH * SEQ * DMODEL * sizeof(_Float16);   // 16.8 MB
    const size_t SZ_W = (size_t)DMODEL * DMODEL * sizeof(_Float16);        // 2.1 MB
    _Float16* Xh  = (_Float16*)ws;              ws += SZ_X;
    _Float16* WqT = (_Float16*)ws;              ws += SZ_W;
    _Float16* WkT = (_Float16*)ws;              ws += SZ_W;
    _Float16* WvT = (_Float16*)ws;              ws += SZ_W;
    _Float16* WoT = (_Float16*)ws;              ws += SZ_W;
    _Float16* Qh  = (_Float16*)ws;              ws += SZ_X;
    _Float16* Kh  = (_Float16*)ws;              ws += SZ_X;
    _Float16* VTh = (_Float16*)ws;              ws += SZ_X;
    _Float16* ctx = (_Float16*)ws;              ws += SZ_X;

    const int NX = BATCH * SEQ * DMODEL;                    // 8388608

    // 1) fp32 -> f16 conversions / weight transposes
    mha_cvt_x<<<dim3(NX / 256), dim3(256), 0, stream>>>(x, Xh, NX);
    mha_cvt_wT<<<dim3((DMODEL * DMODEL) / 256, 4), dim3(256), 0, stream>>>(
        Wq, Wk, Wv, Wo, WqT, WkT, WvT, WoT);

    // 2) fused QKV projection: 128x64 tiles, async-staged B, z = Q/K/V
    mha_gemm<0><<<dim3((BATCH * SEQ) / 128, DMODEL / 64, 3), dim3(256), 0, stream>>>(
        Xh, WqT, WkT, WvT, bq, bk, bv, Qh, Kh, VTh, nullptr);

    // 3) RoPE on Q and K (y selects tensor)
    mha_rope<<<dim3((BATCH * NHEAD * SEQ * (HDIM / 2)) / 256, 2), dim3(256), 0, stream>>>(Qh, Kh);

    // 4) causal flash attention, one wave per (query-tile, b*h)
    mha_attn<<<dim3(SEQ / 16, BATCH * NHEAD), dim3(32), 0, stream>>>(Qh, Kh, VTh, ctx);

    // 5) output projection -> fp32
    mha_gemm<1><<<dim3((BATCH * SEQ) / 128, DMODEL / 64), dim3(256), 0, stream>>>(
        ctx, WoT, nullptr, nullptr, bo, nullptr, nullptr,
        nullptr, nullptr, nullptr, out);
}
